// PosteriorRotationN_446676599075
// MI455X (gfx1250) — matrix operand order repaired and checked by reading, not compiled, not run on hardware
//
#include <hip/hip_runtime.h>
#include <hip/hip_bf16.h>

typedef __attribute__((ext_vector_type(2)))  float    v2f;
typedef __attribute__((ext_vector_type(8)))  float    v8f;
typedef __attribute__((ext_vector_type(16))) _Float16 v16h;
typedef __attribute__((ext_vector_type(2)))  _Float16 v2h;
typedef unsigned int u32;

#define NJ    22
#define HH    8
#define DM    512
#define DEPTH 64
#define CP    2016
#define BB    4
#define TT    100
#define EPSF  0.1f
#define TANHS 10.0f

// f16 WMMA operand loaders per ISA 7.12.2 16-bit layouts.
// A (16xK tile, lane = M row): dwords 0-3 = K [k0+8*half, +8), dwords 4-7 = K [k0+16+8*half, +8)
__device__ __forceinline__ v16h ldA16(const _Float16* base, int k0, int half) {
  union { v16h h; uint4 q[2]; } u;
  const _Float16* p = base + k0 + 8 * half;
  u.q[0] = *(const uint4*)(p);
  u.q[1] = *(const uint4*)(p + 16);
  return u.h;
}
// B (Kx16 tile, lane = N col): K [k0+8*half, +8) then K [k0+16+8*half, +8)
__device__ __forceinline__ v16h ldB16(const _Float16* base, int k0, int half) {
  union { v16h h; uint4 q[2]; } u;
  const _Float16* p = base + k0 + 8 * half;
  u.q[0] = *(const uint4*)(p);
  u.q[1] = *(const uint4*)(p + 16);
  return u.h;
}
// A from LDS packed-dword rows (stride 36 dwords, 16B aligned)
__device__ __forceinline__ v16h ldA16_lds(const u32* row, int k0, int half) {
  union { v16h h; uint4 q[2]; } u;
  int i0 = (k0 >> 1) + 4 * half;
  u.q[0] = *(const uint4*)(row + i0);
  u.q[1] = *(const uint4*)(row + i0 + 8);
  return u.h;
}

#define WMMA16(ACC, A, Bm) \
  (ACC) = __builtin_amdgcn_wmma_f32_16x16x32_f16(false, (A), false, (Bm), (short)0, (ACC), false, false)

// hardware tanh (CDNA5 lists TANH as a transcendental); exp-based fallback
__device__ __forceinline__ float fast_tanh(float x) {
#if __has_builtin(__builtin_amdgcn_tanhf)
  return __builtin_amdgcn_tanhf(x);
#else
  float e = __expf(2.0f * x);
  return 1.0f - 2.0f / (e + 1.0f);
#endif
}

// ---------------------------------------------------------------------------
// Kernel 0: split f32 -> (hi, lo) f16 pair arrays.  x = hi + lo to ~22 bits.
// ---------------------------------------------------------------------------
__global__ __launch_bounds__(256) void cvt_split_kernel(const float* __restrict__ in,
                                                        _Float16* __restrict__ hi,
                                                        _Float16* __restrict__ lo,
                                                        int npair) {
  for (int idx = blockIdx.x * 256 + threadIdx.x; idx < npair; idx += gridDim.x * 256) {
    float2 v = ((const float2*)in)[idx];
    _Float16 h0 = (_Float16)v.x, h1 = (_Float16)v.y;
    v2h H, L;
    H.x = h0;                          H.y = h1;
    L.x = (_Float16)(v.x - (float)h0); L.y = (_Float16)(v.y - (float)h1);
    ((v2h*)hi)[idx] = H;
    ((v2h*)lo)[idx] = L;
  }
}

// ---------------------------------------------------------------------------
// Kernel 1: kv = context @ Wkv^T + bkv (split-f16, f32 out in [B,H,C,d]).
// ---------------------------------------------------------------------------
__global__ __launch_bounds__(128) void kv_proj_kernel(const _Float16* __restrict__ ctxh,
                                                      const _Float16* __restrict__ ctxl,
                                                      const _Float16* __restrict__ wkvh,
                                                      const _Float16* __restrict__ wkvl,
                                                      const float* __restrict__ bkv,
                                                      float* __restrict__ kvp) {
  const int tid  = threadIdx.x;
  const int wid  = blockIdx.x * 4 + (tid >> 5);   // 0..2015
  const int lane = tid & 31;
  const int half = lane >> 4, l15 = lane & 15;
  const int mt2 = wid >> 3;                       // 0..251
  const int ns  = wid & 7;                        // 0..7
  const int m0  = mt2 * 32;
  const int e0  = ns * 64;

  const _Float16* Ah0 = ctxh + (size_t)(m0 + l15) * DM;
  const _Float16* Al0 = ctxl + (size_t)(m0 + l15) * DM;
  const _Float16* Ah1 = Ah0 + (size_t)16 * DM;
  const _Float16* Al1 = Al0 + (size_t)16 * DM;
  const _Float16* Bh  = wkvh + (size_t)(e0 + l15) * DM;
  const _Float16* Bl  = wkvl + (size_t)(e0 + l15) * DM;

  v8f acc[8] = {};
  for (int k0 = 0; k0 < DM; k0 += 32) {
    v16h a0h = ldA16(Ah0, k0, half), a0l = ldA16(Al0, k0, half);
    v16h a1h = ldA16(Ah1, k0, half), a1l = ldA16(Al1, k0, half);
#pragma unroll
    for (int ni = 0; ni < 4; ++ni) {
      v16h bh = ldB16(Bh + (size_t)ni * 16 * DM, k0, half);
      v16h bl = ldB16(Bl + (size_t)ni * 16 * DM, k0, half);
      WMMA16(acc[ni], a0h, bh);
      WMMA16(acc[ni], a0h, bl);
      WMMA16(acc[ni], a0l, bh);
      WMMA16(acc[4 + ni], a1h, bh);
      WMMA16(acc[4 + ni], a1h, bl);
      WMMA16(acc[4 + ni], a1l, bh);
    }
  }

#pragma unroll
  for (int ni = 0; ni < 4; ++ni) {
    const int e = e0 + ni * 16 + l15;
    const int h = e >> 6, dd = e & 63;
    const float bias = bkv[e];
#pragma unroll
    for (int mi = 0; mi < 2; ++mi) {
#pragma unroll
      for (int r = 0; r < 8; ++r) {
        int mm = m0 + mi * 16 + r + 8 * half;
        int bidx = mm / CP, c = mm % CP;
        kvp[(((size_t)bidx * HH + h) * CP + c) * DEPTH + dd] = acc[mi * 4 + ni][r] + bias;
      }
    }
  }
}

// ---------------------------------------------------------------------------
// Kernel 2: LayerNorm over d=64 of kvp [B,H,C,d]; emits hi/lo f16 kv arrays.
// ---------------------------------------------------------------------------
__global__ __launch_bounds__(256) void ln_kernel(const float* __restrict__ kvp,
                                                 const float* __restrict__ gamma,
                                                 const float* __restrict__ beta,
                                                 _Float16* __restrict__ kvh,
                                                 _Float16* __restrict__ kvl) {
  const int tid  = threadIdx.x;
  const int wave = tid >> 5, lane = tid & 31;
  const size_t row = (size_t)blockIdx.x * 8 + wave;   // B*H*C rows
  const float* p = kvp + row * DEPTH;
  float v0 = p[lane * 2], v1 = p[lane * 2 + 1];
  float s  = v0 + v1;
  float sq = v0 * v0 + v1 * v1;
#pragma unroll
  for (int m = 16; m >= 1; m >>= 1) {
    s  += __shfl_xor(s,  m, 32);
    sq += __shfl_xor(sq, m, 32);
  }
  float mu  = s * (1.0f / 64.0f);
  float var = sq * (1.0f / 64.0f) - mu * mu;
  float rs  = rsqrtf(var + 1e-5f);
  float y0 = gamma[lane * 2]     * (v0 - mu) * rs + beta[lane * 2];
  float y1 = gamma[lane * 2 + 1] * (v1 - mu) * rs + beta[lane * 2 + 1];
  _Float16 h0 = (_Float16)y0, h1 = (_Float16)y1;
  v2h H, L;
  H.x = h0;                         H.y = h1;
  L.x = (_Float16)(y0 - (float)h0); L.y = (_Float16)(y1 - (float)h1);
  ((v2h*)(kvh + row * DEPTH))[lane] = H;
  ((v2h*)(kvl + row * DEPTH))[lane] = L;
}

// ---------------------------------------------------------------------------
// Kernel 3: per-joint q projection (split-f16); emits q hi/lo in [N,B,H,T,d].
// ---------------------------------------------------------------------------
__global__ __launch_bounds__(128) void q_proj_kernel(const _Float16* __restrict__ xh,
                                                     const _Float16* __restrict__ xl,
                                                     const _Float16* __restrict__ wqh,
                                                     const _Float16* __restrict__ wql,
                                                     const float* __restrict__ bq,
                                                     _Float16* __restrict__ qh,
                                                     _Float16* __restrict__ ql) {
  const int tid  = threadIdx.x;
  const int wid  = blockIdx.x * 4 + (tid >> 5);   // 0..103
  const int lane = tid & 31;
  const int half = lane >> 4, l15 = lane & 15;
  const int n   = blockIdx.y;
  const int mt2 = wid >> 3;                       // 0..12
  const int ns  = wid & 7;                        // 0..7
  const int m0  = mt2 * 32;
  const int e0  = ns * 64;

  const int mr0 = min(m0 + l15, 399);             // clamp OOB rows; stores guarded
  const int mr1 = min(m0 + 16 + l15, 399);
  const _Float16* Ah0 = xh + ((size_t)mr0 * NJ + n) * DM;
  const _Float16* Al0 = xl + ((size_t)mr0 * NJ + n) * DM;
  const _Float16* Ah1 = xh + ((size_t)mr1 * NJ + n) * DM;
  const _Float16* Al1 = xl + ((size_t)mr1 * NJ + n) * DM;
  const _Float16* Bh  = wqh + ((size_t)n * DM + e0 + l15) * DM;
  const _Float16* Bl  = wql + ((size_t)n * DM + e0 + l15) * DM;

  v8f acc[8] = {};
  for (int k0 = 0; k0 < DM; k0 += 32) {
    v16h a0h = ldA16(Ah0, k0, half), a0l = ldA16(Al0, k0, half);
    v16h a1h = ldA16(Ah1, k0, half), a1l = ldA16(Al1, k0, half);
#pragma unroll
    for (int ni = 0; ni < 4; ++ni) {
      v16h bh = ldB16(Bh + (size_t)ni * 16 * DM, k0, half);
      v16h bl = ldB16(Bl + (size_t)ni * 16 * DM, k0, half);
      WMMA16(acc[ni], a0h, bh);
      WMMA16(acc[ni], a0h, bl);
      WMMA16(acc[ni], a0l, bh);
      WMMA16(acc[4 + ni], a1h, bh);
      WMMA16(acc[4 + ni], a1h, bl);
      WMMA16(acc[4 + ni], a1l, bh);
    }
  }

#pragma unroll
  for (int ni = 0; ni < 4; ++ni) {
    const int e = e0 + ni * 16 + l15;
    const int h = e >> 6, dd = e & 63;
    const float bias = bq[n * DM + e];
#pragma unroll
    for (int mi = 0; mi < 2; ++mi) {
#pragma unroll
      for (int r = 0; r < 8; ++r) {
        int mm = m0 + mi * 16 + r + 8 * half;
        if (mm < 400) {
          int bidx = mm / TT, t = mm % TT;
          size_t idx = ((((size_t)n * BB + bidx) * HH + h) * TT + t) * DEPTH + dd;
          float val = acc[mi * 4 + ni][r] + bias;
          _Float16 vh = (_Float16)val;
          qh[idx] = vh;
          ql[idx] = (_Float16)(val - (float)vh);
        }
      }
    }
  }
}

// ---------------------------------------------------------------------------
// Kernel 4 (fused): one block per (n, b*H+h). q hi/lo + r + Av resident in
// LDS. WMMA tile columns use the PERMUTED pair map c = ct + 126*l15: spacing
// 126 exceeds the max i-run length (63), so the 16 lanes of a tile always
// target 16 distinct Av[t][i] words -> conflict-free LDS atomics.
// ---------------------------------------------------------------------------
__global__ __launch_bounds__(256) void skew_rotate_kernel(const _Float16* __restrict__ qh,
                                                          const _Float16* __restrict__ ql,
                                                          const _Float16* __restrict__ kvh,
                                                          const _Float16* __restrict__ kvl,
                                                          const float* __restrict__ x,
                                                          float* __restrict__ out) {
  __shared__ u32  qh_s[112 * 36];   // packed 2xf16 per dword; stride 36 (bank-safe, 16B aligned)
  __shared__ u32  ql_s[112 * 36];
  __shared__ float r_s[112 * 65];
  __shared__ float Av [112 * 65];

  const int tid = threadIdx.x;
  const int n   = blockIdx.x;       // 0..21
  const int zh  = blockIdx.y;       // b*8 + h
  const int b = zh >> 3, h = zh & 7;

  const u32* qhb = (const u32*)(qh + ((size_t)n * 32 + zh) * TT * DEPTH);
  const u32* qlb = (const u32*)(ql + ((size_t)n * 32 + zh) * TT * DEPTH);
  const _Float16* kvhb = kvh + (size_t)zh * CP * DEPTH;
  const _Float16* kvlb = kvl + (size_t)zh * CP * DEPTH;

  // stage q hi/lo (dwords); rows t>=100 zero (self-masking: s=0)
  for (int idx = tid; idx < 112 * 32; idx += 256) {
    int rowl = idx >> 5, cw = idx & 31;
    u32 vh = 0, vl = 0;
    if (rowl < TT) {
      vh = qhb[rowl * 32 + cw];
      vl = qlb[rowl * 32 + cw];
    }
    qh_s[rowl * 36 + cw] = vh;
    ql_s[rowl * 36 + cw] = vl;
  }
  // stage r (f32), zero Av
  for (int idx = tid; idx < 112 * 64; idx += 256) {
    int rowl = idx >> 6, col = idx & 63;
    float rv = 0.f;
    if (rowl < TT)
      rv = x[(((size_t)b * TT + rowl) * NJ + n) * DM + h * DEPTH + col];
    r_s[rowl * 65 + col] = rv;
    Av [rowl * 65 + col] = 0.f;
  }
  __syncthreads();

  const int wave = tid >> 5, lane = tid & 31;
  const int half = lane >> 4, l15 = lane & 15;

  for (int ct = wave; ct < CP / 16; ct += 8) {
    const int c = ct + 126 * l15;   // permuted column<->pair map (bijective on [0,2016))

    // cache this c-tile's kv B-operands (hi/lo, both K=32 chunks) in registers
    const _Float16* kvrh = kvhb + (size_t)c * DEPTH;
    const _Float16* kvrl = kvlb + (size_t)c * DEPTH;
    v16h bh0 = ldB16(kvrh, 0, half),  bl0 = ldB16(kvrl, 0, half);
    v16h bh1 = ldB16(kvrh, 32, half), bl1 = ldB16(kvrl, 32, half);

    // decode pair index c -> (i, j), i > j, c = i*(i-1)/2 + j
    int i = (int)floorf((1.0f + sqrtf(1.0f + 8.0f * (float)c)) * 0.5f);
    while (i * (i - 1) / 2 > c) --i;
    while ((i + 1) * i / 2 <= c) ++i;
    int j = c - i * (i - 1) / 2;

    for (int mt2 = 0; mt2 < 7; ++mt2) {
      const u32* rh = qh_s + (mt2 * 16 + l15) * 36;
      const u32* rl = ql_s + (mt2 * 16 + l15) * 36;
      v8f acc = {};
      v16h ah = ldA16_lds(rh, 0, half), al = ldA16_lds(rl, 0, half);
      WMMA16(acc, ah, bh0);
      WMMA16(acc, ah, bl0);
      WMMA16(acc, al, bh0);
      ah = ldA16_lds(rh, 32, half); al = ldA16_lds(rl, 32, half);
      WMMA16(acc, ah, bh1);
      WMMA16(acc, ah, bl1);
      WMMA16(acc, al, bh1);

#pragma unroll
      for (int r = 0; r < 8; ++r) {
        int tl = mt2 * 16 + r + 8 * half;
        float sv = TANHS * fast_tanh(acc[r] * (1.0f / TANHS));
        atomicAdd(&Av[tl * 65 + i],  sv * r_s[tl * 65 + j]);
        atomicAdd(&Av[tl * 65 + j], -sv * r_s[tl * 65 + i]);
      }
    }
  }
  __syncthreads();

  for (int idx = tid; idx < 112 * 64; idx += 256) {
    int rowl = idx >> 6, col = idx & 63;
    if (rowl < TT) {
      out[(((size_t)b * TT + rowl) * NJ + n) * DM + h * DEPTH + col] =
          r_s[rowl * 65 + col] + EPSF * Av[rowl * 65 + col];
    }
  }
}

// ---------------------------------------------------------------------------
extern "C" void kernel_launch(void* const* d_in, const int* in_sizes, int n_in,
                              void* d_out, int out_size, void* d_ws, size_t ws_size,
                              hipStream_t stream) {
  const float* x       = (const float*)d_in[0];  // [4,100,22,512]
  const float* context = (const float*)d_in[1];  // [4,2016,512]
  const float* Wq      = (const float*)d_in[2];  // [22,512,512]
  const float* bq      = (const float*)d_in[3];  // [22,512]
  const float* Wkv     = (const float*)d_in[4];  // [512,512]
  const float* bkv     = (const float*)d_in[5];  // [512]
  const float* gamma   = (const float*)d_in[6];  // [64]
  const float* beta    = (const float*)d_in[7];  // [64]
  float* out = (float*)d_out;                    // [4,100,22,512]

  const size_t nCTX = (size_t)BB * CP * DM;              // 4,128,768
  const size_t nWKV = (size_t)DM * DM;                   //   262,144
  const size_t nX   = (size_t)BB * TT * NJ * DM;         // 4,505,600
  const size_t nWQ  = (size_t)NJ * DM * DM;              // 5,767,168
  const size_t nKV  = (size_t)BB * HH * CP * DEPTH;      // 4,128,768
  const size_t nQ   = (size_t)NJ * BB * HH * TT * DEPTH; // 4,505,600

  char* w = (char*)d_ws;
  float*    kvp  = (float*)w;     w += nKV * 4;
  _Float16* ctxh = (_Float16*)w;  w += nCTX * 2;
  _Float16* ctxl = (_Float16*)w;  w += nCTX * 2;
  _Float16* wkvh = (_Float16*)w;  w += nWKV * 2;
  _Float16* wkvl = (_Float16*)w;  w += nWKV * 2;
  _Float16* xh   = (_Float16*)w;  w += nX * 2;
  _Float16* xl   = (_Float16*)w;  w += nX * 2;
  _Float16* wqh  = (_Float16*)w;  w += nWQ * 2;
  _Float16* wql  = (_Float16*)w;  w += nWQ * 2;
  _Float16* kvhp = (_Float16*)w;  w += nKV * 2;
  _Float16* kvlp = (_Float16*)w;  w += nKV * 2;
  _Float16* qhp  = (_Float16*)w;  w += nQ * 2;
  _Float16* qlp  = (_Float16*)w;  /* total ~110 MB */

  auto cvtGrid = [](size_t npair) { return (unsigned)((npair + 2047) / 2048); };

  // 0) split-f16 conversions (memory-bound streams)
  cvt_split_kernel<<<cvtGrid(nCTX / 2), 256, 0, stream>>>(context, ctxh, ctxl, (int)(nCTX / 2));
  cvt_split_kernel<<<cvtGrid(nWKV / 2), 256, 0, stream>>>(Wkv, wkvh, wkvl, (int)(nWKV / 2));
  cvt_split_kernel<<<cvtGrid(nX   / 2), 256, 0, stream>>>(x, xh, xl, (int)(nX / 2));
  cvt_split_kernel<<<cvtGrid(nWQ  / 2), 256, 0, stream>>>(Wq, wqh, wql, (int)(nWQ / 2));
  // 1) kv projection (+bias) -> f32 kvp in [B,H,C,d]
  kv_proj_kernel<<<dim3(504), 128, 0, stream>>>(ctxh, ctxl, wkvh, wkvl, bkv, kvp);
  // 2) LayerNorm -> kv hi/lo f16
  ln_kernel<<<dim3((BB * HH * CP) / 8), 256, 0, stream>>>(kvp, gamma, beta, kvhp, kvlp);
  // 3) per-joint q projection -> q hi/lo f16
  q_proj_kernel<<<dim3(26, NJ), 128, 0, stream>>>(xh, xl, wqh, wql, bq, qhp, qlp);
  // 4) fused skew scores + antisymmetric rotation
  skew_rotate_kernel<<<dim3(NJ, BB * HH), 256, 0, stream>>>(qhp, qlp, kvhp, kvlp, x, out);
}
